// CGCNN_CALPHAD_23124103921773
// MI455X (gfx1250) — matrix operand order repaired and checked by reading, not compiled, not run on hardware
//
#include <hip/hip_runtime.h>
#include <hip/hip_bf16.h>

#define ND 64
#define ED 32
#define HD 128
#define EPSBN 1e-5f

typedef __attribute__((ext_vector_type(16))) _Float16 v16h;
typedef __attribute__((ext_vector_type(8)))  float    v8f;

union Frag16 {
  v16h v;
  uint4 u[2];
  _Float16 h[16];
};

__device__ __forceinline__ float softplus_f(float x) {
  // stable softplus: max(x,0) + log1p(exp(-|x|))
  float ax = fabsf(x);
  return fmaxf(x, 0.0f) + log1pf(__expf(-ax));
}

// A-fragment (16x32 f16, row M = lane%16). Per ISA layout, lane holds
// K = 8*half + [0..7] (elems 0-7) and K = 16 + 8*half + [0..7] (elems 8-15).
__device__ __forceinline__ v16h load_a_h16(const _Float16* base, int half) {
  Frag16 f;
  f.u[0] = *(const uint4*)(base + 8 * half);
  f.u[1] = *(const uint4*)(base + 16 + 8 * half);
  return f.v;
}

__device__ __forceinline__ v16h load_a_f32(const float* base, int half) {
  const float* p0 = base + 8 * half;
  const float* p1 = base + 16 + 8 * half;
  float4 a0 = *(const float4*)(p0);
  float4 a1 = *(const float4*)(p0 + 4);
  float4 b0 = *(const float4*)(p1);
  float4 b1 = *(const float4*)(p1 + 4);
  Frag16 f;
  f.h[0]  = (_Float16)a0.x; f.h[1]  = (_Float16)a0.y;
  f.h[2]  = (_Float16)a0.z; f.h[3]  = (_Float16)a0.w;
  f.h[4]  = (_Float16)a1.x; f.h[5]  = (_Float16)a1.y;
  f.h[6]  = (_Float16)a1.z; f.h[7]  = (_Float16)a1.w;
  f.h[8]  = (_Float16)b0.x; f.h[9]  = (_Float16)b0.y;
  f.h[10] = (_Float16)b0.z; f.h[11] = (_Float16)b0.w;
  f.h[12] = (_Float16)b1.x; f.h[13] = (_Float16)b1.y;
  f.h[14] = (_Float16)b1.z; f.h[15] = (_Float16)b1.w;
  return f.v;
}

// B-fragment (32x16 f16, col N = lane%16). Lane holds K = 8*half + [0..7]
// (elems 0-7) and K = 16 + 8*half + [0..7] (elems 8-15), matching A.
__device__ __forceinline__ v16h load_b_lds(const _Float16* colbase, int half) {
  Frag16 f;
  f.u[0] = *(const uint4*)(colbase + 8 * half);
  f.u[1] = *(const uint4*)(colbase + 16 + 8 * half);
  return f.v;
}

__device__ __forceinline__ v8f wmma16(v16h a, v16h b, v8f c) {
  return __builtin_amdgcn_wmma_f32_16x16x32_f16(false, a, false, b, (short)0, c,
                                                false, false);
}

// ---------------------------------------------------------------- node proj
__global__ __launch_bounds__(128) void k_node_proj(
    const float* __restrict__ x, const float* __restrict__ Wnp,
    const float* __restrict__ bnp, float* __restrict__ hraw,
    float* __restrict__ bnsum, float* __restrict__ bnsq, int N) {
  __shared__ float ssum[ND];
  __shared__ float ssq[ND];
  int tid = threadIdx.x;
  if (tid < ND) { ssum[tid] = 0.0f; ssq[tid] = 0.0f; }
  __syncthreads();
  int n = blockIdx.x * 128 + tid;
  if (n < N) {
    float xv[13];
#pragma unroll
    for (int i = 0; i < 13; i++) xv[i] = x[(size_t)n * 13 + i];
    for (int c = 0; c < ND; c++) {
      float a = bnp[c];
#pragma unroll
      for (int i = 0; i < 13; i++) a += xv[i] * Wnp[i * ND + c];
      float s = softplus_f(a);
      hraw[(size_t)n * ND + c] = s;
      atomicAdd(&ssum[c], s);
      atomicAdd(&ssq[c], s * s);
    }
  }
  __syncthreads();
  if (tid < ND) {
    atomicAdd(&bnsum[tid], ssum[tid]);
    atomicAdd(&bnsq[tid], ssq[tid]);
  }
}

// ---------------------------------------------------------------- BN apply
__global__ void k_bn_apply(const float* __restrict__ srcb,
                           const float* __restrict__ gamma,
                           const float* __restrict__ beta,
                           const float* __restrict__ bnsum,
                           const float* __restrict__ bnsq,
                           float* __restrict__ hf, _Float16* __restrict__ h16,
                           int N) {
  size_t idx = (size_t)blockIdx.x * blockDim.x + threadIdx.x;
  if (idx >= (size_t)N * ND) return;
  int c = (int)(idx & (ND - 1));
  float invN = 1.0f / (float)N;
  float mu = bnsum[c] * invN;
  float var = bnsq[c] * invN - mu * mu;
  float y = (srcb[idx] - mu) * rsqrtf(var + EPSBN) * gamma[c] + beta[c];
  hf[idx] = y;
  h16[idx] = (_Float16)y;
}

// ---------------------------------------------------------------- edge proj
__global__ void k_edge_proj(const float* __restrict__ ea,
                            const float* __restrict__ Wep,
                            const float* __restrict__ bep,
                            _Float16* __restrict__ e16, int E) {
  int e = blockIdx.x * blockDim.x + threadIdx.x;
  if (e >= E) return;
  float a0 = ea[(size_t)e * 2 + 0];
  float a1 = ea[(size_t)e * 2 + 1];
#pragma unroll
  for (int c = 0; c < ED; c++) {
    float s = softplus_f(a0 * Wep[c] + a1 * Wep[ED + c] + bep[c]);
    e16[(size_t)e * ED + c] = (_Float16)s;
  }
}

// ---------------------------------------------------------------- edge conv
__global__ __launch_bounds__(256) void k_edge_conv(
    const _Float16* __restrict__ h16, const _Float16* __restrict__ e16,
    const int* __restrict__ src, const int* __restrict__ dst,
    const float* __restrict__ We1, const float* __restrict__ be1,
    const float* __restrict__ We2, const float* __restrict__ be2,
    float* __restrict__ aggr, int E, int numTiles) {
  extern __shared__ __align__(16) unsigned char smem_raw[];
  _Float16* W1T = (_Float16*)smem_raw;      // [128 cols][160 k]
  _Float16* W2T = W1T + HD * 160;           // [128 cols][128 k]
  _Float16* tstage = W2T + HD * HD;         // [8 waves][16*128]

  const int tid = threadIdx.x;
  for (int i = tid; i < HD * 160; i += 256) {
    int col = i / 160, k = i - col * 160;
    W1T[i] = (_Float16)We1[k * HD + col];
  }
  for (int i = tid; i < HD * HD; i += 256) {
    int col = i >> 7, k = i & 127;
    W2T[i] = (_Float16)We2[k * HD + col];
  }
  __syncthreads();

  const int wave = tid >> 5;
  const int lane = tid & 31;
  const int lane16 = lane & 15;
  const int half = lane >> 4;
  _Float16* tst = tstage + wave * (16 * HD);

  for (int tile = blockIdx.x * 8 + wave; tile < numTiles;
       tile += gridDim.x * 8) {
    int erow = tile * 16 + lane16;
    int ec = erow < E ? erow : 0;
    int sI = src[ec], dI = dst[ec];
    const _Float16* baseD = h16 + (size_t)dI * ND;
    const _Float16* baseS = h16 + (size_t)sI * ND;
    const _Float16* baseE = e16 + (size_t)ec * ED;

    v8f acc[8];
#pragma unroll
    for (int n = 0; n < 8; n++) {
      float b = be1[n * 16 + lane16];
#pragma unroll
      for (int j = 0; j < 8; j++) acc[n][j] = b;
    }
#pragma unroll
    for (int c = 0; c < 5; c++) {
      const _Float16* ab =
          (c < 2) ? (baseD + c * 32) : (c < 4) ? (baseS + (c - 2) * 32) : baseE;
      v16h a = load_a_h16(ab, half);
#pragma unroll
      for (int n = 0; n < 8; n++) {
        v16h b = load_b_lds(W1T + (n * 16 + lane16) * 160 + c * 32, half);
        acc[n] = wmma16(a, b, acc[n]);
      }
    }
    // softplus + restage (C layout -> row-major tile for next A-fragments)
#pragma unroll
    for (int n = 0; n < 8; n++) {
#pragma unroll
      for (int v = 0; v < 8; v++) {
        tst[(v + 8 * half) * HD + n * 16 + lane16] =
            (_Float16)softplus_f(acc[n][v]);
      }
    }
    asm volatile("s_wait_dscnt 0" ::: "memory");

    v8f acc2[8];
#pragma unroll
    for (int n = 0; n < 8; n++) {
      float b = be2[n * 16 + lane16];
#pragma unroll
      for (int j = 0; j < 8; j++) acc2[n][j] = b;
    }
#pragma unroll
    for (int c = 0; c < 4; c++) {
      v16h a = load_a_h16(tst + lane16 * HD + c * 32, half);
#pragma unroll
      for (int n = 0; n < 8; n++) {
        v16h b = load_b_lds(W2T + (n * 16 + lane16) * HD + c * 32, half);
        acc2[n] = wmma16(a, b, acc2[n]);
      }
    }
    // softplus + scatter into aggr[dst] (segment_sum)
#pragma unroll
    for (int v = 0; v < 8; v++) {
      int er = tile * 16 + v + 8 * half;
      if (er < E) {
        int dd = dst[er];
        float* ap = aggr + (size_t)dd * HD + lane16;
#pragma unroll
        for (int n = 0; n < 8; n++)
          atomicAdd(ap + n * 16, softplus_f(acc2[n][v]));
      }
    }
    asm volatile("s_wait_dscnt 0" ::: "memory");
  }
}

// ---------------------------------------------------------------- node conv
__global__ __launch_bounds__(256) void k_node_conv(
    const _Float16* __restrict__ h16, const float* __restrict__ hf,
    const float* __restrict__ aggr, const float* __restrict__ Wn1,
    const float* __restrict__ bn1, const float* __restrict__ Wn2,
    const float* __restrict__ bn2, float* __restrict__ hraw,
    float* __restrict__ bnsum, float* __restrict__ bnsq, int N, int numTiles) {
  extern __shared__ __align__(16) unsigned char smem_raw2[];
  _Float16* W1T = (_Float16*)smem_raw2;  // [128 cols][192 k]
  _Float16* W2T = W1T + HD * 192;        // [64 cols][128 k]
  _Float16* tstage = W2T + ND * HD;      // [8 waves][16*128]

  const int tid = threadIdx.x;
  for (int i = tid; i < HD * 192; i += 256) {
    int col = i / 192, k = i - col * 192;
    W1T[i] = (_Float16)Wn1[k * HD + col];
  }
  for (int i = tid; i < ND * HD; i += 256) {
    int col = i >> 7, k = i & 127;
    W2T[i] = (_Float16)Wn2[k * ND + col];
  }
  __syncthreads();

  const int wave = tid >> 5;
  const int lane = tid & 31;
  const int lane16 = lane & 15;
  const int half = lane >> 4;
  _Float16* tst = tstage + wave * (16 * HD);

  for (int tile = blockIdx.x * 8 + wave; tile < numTiles;
       tile += gridDim.x * 8) {
    int nrow = tile * 16 + lane16;
    int nc = nrow < N ? nrow : 0;
    const _Float16* baseH = h16 + (size_t)nc * ND;
    const float* baseA = aggr + (size_t)nc * HD;

    v8f acc[8];
#pragma unroll
    for (int n = 0; n < 8; n++) {
      float b = bn1[n * 16 + lane16];
#pragma unroll
      for (int j = 0; j < 8; j++) acc[n][j] = b;
    }
#pragma unroll
    for (int c = 0; c < 6; c++) {
      v16h a = (c < 2) ? load_a_h16(baseH + c * 32, half)
                       : load_a_f32(baseA + (c - 2) * 32, half);
#pragma unroll
      for (int n = 0; n < 8; n++) {
        v16h b = load_b_lds(W1T + (n * 16 + lane16) * 192 + c * 32, half);
        acc[n] = wmma16(a, b, acc[n]);
      }
    }
#pragma unroll
    for (int n = 0; n < 8; n++) {
#pragma unroll
      for (int v = 0; v < 8; v++) {
        tst[(v + 8 * half) * HD + n * 16 + lane16] =
            (_Float16)softplus_f(acc[n][v]);
      }
    }
    asm volatile("s_wait_dscnt 0" ::: "memory");

    v8f acc2[4];
#pragma unroll
    for (int n = 0; n < 4; n++) {
      float b = bn2[n * 16 + lane16];
#pragma unroll
      for (int j = 0; j < 8; j++) acc2[n][j] = b;
    }
#pragma unroll
    for (int c = 0; c < 4; c++) {
      v16h a = load_a_h16(tst + lane16 * HD + c * 32, half);
#pragma unroll
      for (int n = 0; n < 4; n++) {
        v16h b = load_b_lds(W2T + (n * 16 + lane16) * HD + c * 32, half);
        acc2[n] = wmma16(a, b, acc2[n]);
      }
    }
    // residual add + store raw + BN partial sums
    float sums[4] = {0.f, 0.f, 0.f, 0.f};
    float sqs[4] = {0.f, 0.f, 0.f, 0.f};
#pragma unroll
    for (int v = 0; v < 8; v++) {
      int nr = tile * 16 + v + 8 * half;
      if (nr < N) {
        float* hp = hraw + (size_t)nr * ND + lane16;
        const float* hold = hf + (size_t)nr * ND + lane16;
#pragma unroll
        for (int n = 0; n < 4; n++) {
          float val = acc2[n][v] + hold[n * 16];
          hp[n * 16] = val;
          sums[n] += val;
          sqs[n] += val * val;
        }
      }
    }
#pragma unroll
    for (int n = 0; n < 4; n++) {
      atomicAdd(&bnsum[n * 16 + lane16], sums[n]);
      atomicAdd(&bnsq[n * 16 + lane16], sqs[n]);
    }
    asm volatile("s_wait_dscnt 0" ::: "memory");
  }
}

// ---------------------------------------------------------------- pooling
__global__ void k_pool(const float* __restrict__ hf,
                       const int* __restrict__ batch,
                       float* __restrict__ pooled, float* __restrict__ cnt,
                       int N) {
  int n = blockIdx.x * blockDim.x + threadIdx.x;
  if (n >= N) return;
  int b = batch[n];
  atomicAdd(&cnt[b], 1.0f);
  const float* hp = hf + (size_t)n * ND;
  float* pp = pooled + (size_t)b * ND;
#pragma unroll
  for (int c = 0; c < ND; c++) atomicAdd(&pp[c], hp[c]);
}

// ---------------------------------------------------------------- head MLP
__global__ __launch_bounds__(128) void k_head(
    const float* __restrict__ pooled, const float* __restrict__ cnt,
    const float* __restrict__ Wo1, const float* __restrict__ bo1,
    const float* __restrict__ Wo2, const float* __restrict__ bo2,
    float* __restrict__ out, int G) {
  int g = blockIdx.x * blockDim.x + threadIdx.x;
  if (g >= G) return;
  float inv = 1.0f / fmaxf(cnt[g], 1.0f);
  float p[ND];
#pragma unroll
  for (int i = 0; i < ND; i++) p[i] = pooled[(size_t)g * ND + i] * inv;
  float o = bo2[0];
  for (int j = 0; j < HD; j++) {
    float t = bo1[j];
#pragma unroll
    for (int i = 0; i < ND; i++) t += p[i] * Wo1[i * HD + j];
    o += softplus_f(t) * Wo2[j];
  }
  out[g] = o;
}

// ================================================================ launcher
extern "C" void kernel_launch(void* const* d_in, const int* in_sizes, int n_in,
                              void* d_out, int out_size, void* d_ws,
                              size_t ws_size, hipStream_t stream) {
  const float* x = (const float*)d_in[0];
  const float* ea = (const float*)d_in[1];
  const int* eidx = (const int*)d_in[2];
  const int* batch = (const int*)d_in[3];
  const float* Wnp = (const float*)d_in[4];
  const float* bnp = (const float*)d_in[5];
  const float* g_np = (const float*)d_in[6];
  const float* be_np = (const float*)d_in[7];
  const float* Wep = (const float*)d_in[8];
  const float* bep = (const float*)d_in[9];
  const float* We1 = (const float*)d_in[10];
  const float* be1 = (const float*)d_in[11];
  const float* We2 = (const float*)d_in[12];
  const float* be2 = (const float*)d_in[13];
  const float* Wn1 = (const float*)d_in[14];
  const float* bn1 = (const float*)d_in[15];
  const float* Wn2 = (const float*)d_in[16];
  const float* bn2 = (const float*)d_in[17];
  const float* g_bn = (const float*)d_in[18];
  const float* b_bn = (const float*)d_in[19];
  const float* Wo1 = (const float*)d_in[20];
  const float* bo1 = (const float*)d_in[21];
  const float* Wo2 = (const float*)d_in[22];
  const float* bo2 = (const float*)d_in[23];
  (void)n_in;

  int N = in_sizes[0] / 13;
  int E = in_sizes[1] / 2;
  int G = out_size;
  const int* src = eidx;
  const int* dst = eidx + E;

  unsigned char* ws = (unsigned char*)d_ws;
  size_t off = 0;
  auto take = [&](size_t bytes) -> unsigned char* {
    unsigned char* p = ws + off;
    off += (bytes + 255) & ~(size_t)255;
    return p;
  };
  float* hf = (float*)take((size_t)N * ND * 4);
  _Float16* h16 = (_Float16*)take((size_t)N * ND * 2);
  float* hraw = (float*)take((size_t)N * ND * 4);
  float* aggr = (float*)take((size_t)N * HD * 4);
  _Float16* e16 = (_Float16*)take((size_t)E * ED * 2);
  float* bnsum = (float*)take(ND * 4);
  float* bnsq = (float*)take(ND * 4);
  float* pooled = (float*)take((size_t)G * ND * 4);
  float* cnt = (float*)take((size_t)G * 4);
  (void)ws_size;

  // node projection + BN
  hipMemsetAsync(bnsum, 0, ND * 4, stream);
  hipMemsetAsync(bnsq, 0, ND * 4, stream);
  k_node_proj<<<(N + 127) / 128, 128, 0, stream>>>(x, Wnp, bnp, hraw, bnsum,
                                                   bnsq, N);
  int bnBlocks = (int)(((size_t)N * ND + 255) / 256);
  k_bn_apply<<<bnBlocks, 256, 0, stream>>>(hraw, g_np, be_np, bnsum, bnsq, hf,
                                           h16, N);
  // edge projection
  k_edge_proj<<<(E + 255) / 256, 256, 0, stream>>>(ea, Wep, bep, e16, E);

  int edgeTiles = (E + 15) / 16;
  int nodeTiles = (N + 15) / 16;
  size_t shmE = (size_t)(HD * 160 + HD * HD + 8 * 16 * HD) * 2;
  size_t shmN = (size_t)(HD * 192 + ND * HD + 8 * 16 * HD) * 2;

  for (int l = 0; l < 3; l++) {
    hipMemsetAsync(aggr, 0, (size_t)N * HD * 4, stream);
    k_edge_conv<<<640, 256, shmE, stream>>>(
        h16, e16, src, dst, We1 + (size_t)l * 160 * HD, be1 + l * HD,
        We2 + (size_t)l * HD * HD, be2 + l * HD, aggr, E, edgeTiles);
    hipMemsetAsync(bnsum, 0, ND * 4, stream);
    hipMemsetAsync(bnsq, 0, ND * 4, stream);
    k_node_conv<<<256, 256, shmN, stream>>>(
        h16, hf, aggr, Wn1 + (size_t)l * 192 * HD, bn1 + l * HD,
        Wn2 + (size_t)l * HD * ND, bn2 + l * ND, hraw, bnsum, bnsq, N,
        nodeTiles);
    k_bn_apply<<<bnBlocks, 256, 0, stream>>>(hraw, g_bn + l * ND, b_bn + l * ND,
                                             bnsum, bnsq, hf, h16, N);
  }

  hipMemsetAsync(pooled, 0, (size_t)G * ND * 4, stream);
  hipMemsetAsync(cnt, 0, (size_t)G * 4, stream);
  k_pool<<<(N + 255) / 256, 256, 0, stream>>>(hf, batch, pooled, cnt, N);
  k_head<<<(G + 127) / 128, 128, 0, stream>>>(pooled, cnt, Wo1, bo1, Wo2, bo2,
                                              (float*)d_out, G);
}
